// PromptEncoder_71107478553077
// MI455X (gfx1250) — matrix operand order repaired and checked by reading, not compiled, not run on hardware
//
#include <hip/hip_runtime.h>
#include <hip/hip_bf16.h>
#include <cstdint>
#include <cstddef>

#define E_DIM  1024
#define H_DIM  1024
#define HD     512
#define G4     2048   // 4*HD
#define P_DIM  16
#define B_DIM  32
#define S_DIM  256

typedef __bf16 bf16_t;
typedef bf16_t v16bf __attribute__((ext_vector_type(16)));
typedef bf16_t v8bf  __attribute__((ext_vector_type(8)));
typedef float  v8f   __attribute__((ext_vector_type(8)));

static __device__ __forceinline__ v16bf ld_afrag(const bf16_t* p) {
    // A 16x32 bf16 fragment half: elements 0-15 = K+0..15 (lane<16)
    v8bf a0 = *(const v8bf*)(p);
    v8bf a1 = *(const v8bf*)(p + 8);
    return __builtin_shufflevector(a0, a1, 0,1,2,3,4,5,6,7,8,9,10,11,12,13,14,15);
}

// ---------------------------------------------------------------------------
// small utility kernels
// ---------------------------------------------------------------------------
__global__ void k_fill_i32(int* __restrict__ p, int v, int n) {
    int t = blockIdx.x * blockDim.x + threadIdx.x;
    if (t < n) p[t] = v;
}

__global__ void k_zero_u32(uint32_t* __restrict__ p, int n) {
    int t = blockIdx.x * blockDim.x + threadIdx.x;
    if (t < n) p[t] = 0u;
}

__global__ void k_f32_to_bf16(const float* __restrict__ s, bf16_t* __restrict__ d, int n) {
    int t = blockIdx.x * blockDim.x + threadIdx.x;
    if (t < n) d[t] = (bf16_t)s[t];
}

__global__ void k_combine_bias(const float* __restrict__ a, const float* __restrict__ b,
                               float* __restrict__ o, int n) {
    int t = blockIdx.x * blockDim.x + threadIdx.x;
    if (t < n) o[t] = a[t] + b[t];
}

// map[b*S+s] = prompt slot index p or -1
__global__ void k_scatter_map(const int* __restrict__ pidx, int* __restrict__ map, int n) {
    int t = blockIdx.x * blockDim.x + threadIdx.x;       // over B*P
    if (t < n) {
        int b = t / P_DIM;
        int p = t - b * P_DIM;
        map[b * S_DIM + pidx[t]] = p;
    }
}

// raw -> d_out (f32, [B,S,E]); allb -> bf16 time-major [S,B,E]
__global__ void k_embed(const float* __restrict__ emb, const float* __restrict__ ctx,
                        const float* __restrict__ pemb, const int* __restrict__ ids,
                        const int* __restrict__ map, float* __restrict__ raw_out,
                        bf16_t* __restrict__ allb, long long n) {
    long long tid = (long long)blockIdx.x * blockDim.x + threadIdx.x;
    if (tid >= n) return;
    int e  = (int)(tid & (E_DIM - 1));
    long long bs = tid >> 10;                 // E = 1024
    int s  = (int)(bs & (S_DIM - 1));
    int b  = (int)(bs >> 8);
    int id = ids[bs];
    float raw = emb[(size_t)id * E_DIM + e];
    raw_out[tid] = raw;
    int pm = map[bs];
    float v = (pm >= 0) ? pemb[(size_t)pm * E_DIM + e]
                        : (raw + ctx[(size_t)id * E_DIM + e]);
    allb[((size_t)s * B_DIM + b) * E_DIM + e] = (bf16_t)v;
}

// d_out[b, s=pidx[b,p], e] = mlp_out[(s*B+b)*E+e] + prompt_emb[p,e]
__global__ void k_scatter_out(const float* __restrict__ mlp, const float* __restrict__ pemb,
                              const int* __restrict__ pidx, float* __restrict__ out, int n) {
    int t = blockIdx.x * blockDim.x + threadIdx.x;       // over B*P*E
    if (t >= n) return;
    int e = t & (E_DIM - 1);
    int bp = t >> 10;
    int b = bp / P_DIM;
    int p = bp - b * P_DIM;
    int s = pidx[b * P_DIM + p];
    out[((size_t)b * S_DIM + s) * E_DIM + e] =
        mlp[((size_t)s * B_DIM + b) * E_DIM + e] + pemb[(size_t)p * E_DIM + e];
}

// ---------------------------------------------------------------------------
// bf16 WMMA GEMM, 2x2 register blocking: each wave owns a 32x32 output macro
// tile (4x 16x16 WMMA tiles). Per k-step: 2 A-frags + 2 B-frags -> 4 WMMAs.
// C[M,N] = A[M,K] * Bt[N,K]^T  (+bias, opt relu); M,N multiples of 32.
// ---------------------------------------------------------------------------
__global__ __launch_bounds__(256) void k_gemm_bf16(
    const bf16_t* __restrict__ A, const bf16_t* __restrict__ Bt,
    const float* __restrict__ bias, float* __restrict__ Cf,
    bf16_t* __restrict__ Cb, int M, int N, int K, int relu)
{
    const int lane = threadIdx.x & 31;
    const int wave = threadIdx.x >> 5;
    const int tilesN2 = N >> 5;
    const int totalTiles = (M >> 5) * tilesN2;
    int tile = blockIdx.x * 8 + wave;               // wave-uniform -> EXEC stays full
    if (tile >= totalTiles) return;
    const int mt2 = tile / tilesN2;
    const int nt2 = tile - mt2 * tilesN2;
    const int l15 = lane & 15;
    const int hi  = lane >> 4;

    const bf16_t* Ap0 = A  + (size_t)(mt2 * 32 + l15) * K + hi * 16;
    const bf16_t* Ap1 = Ap0 + (size_t)16 * K;
    const bf16_t* Bp0 = Bt + (size_t)(nt2 * 32 + l15) * K + hi * 16;
    const bf16_t* Bp1 = Bp0 + (size_t)16 * K;

    v8f acc00 = {}, acc01 = {}, acc10 = {}, acc11 = {};
#pragma unroll 2
    for (int k = 0; k < K; k += 32) {
        v16bf a0 = ld_afrag(Ap0 + k);
        v16bf a1 = ld_afrag(Ap1 + k);
        v16bf b0 = *(const v16bf*)(Bp0 + k);
        v16bf b1 = *(const v16bf*)(Bp1 + k);
        acc00 = __builtin_amdgcn_wmma_f32_16x16x32_bf16(false, a0, false, b0, (short)0, acc00, false, false);
        acc01 = __builtin_amdgcn_wmma_f32_16x16x32_bf16(false, a0, false, b1, (short)0, acc01, false, false);
        acc10 = __builtin_amdgcn_wmma_f32_16x16x32_bf16(false, a1, false, b0, (short)0, acc10, false, false);
        acc11 = __builtin_amdgcn_wmma_f32_16x16x32_bf16(false, a1, false, b1, (short)0, acc11, false, false);
    }

    const v8f* accs[4] = { &acc00, &acc01, &acc10, &acc11 };
#pragma unroll
    for (int i = 0; i < 2; ++i) {
#pragma unroll
        for (int j = 0; j < 2; ++j) {
            const v8f& a = *accs[i * 2 + j];
            const int colN = nt2 * 32 + j * 16 + l15;
            const float bb = bias ? bias[colN] : 0.0f;
            const int rowBase = mt2 * 32 + i * 16 + hi * 8;
#pragma unroll
            for (int r = 0; r < 8; ++r) {
                float v = a[r] + bb;
                if (relu) v = v > 0.0f ? v : 0.0f;
                size_t idx = (size_t)(rowBase + r) * N + colN;
                if (Cf) Cf[idx] = v;
                if (Cb) Cb[idx] = (bf16_t)v;
            }
        }
    }
}

// ---------------------------------------------------------------------------
// one LSTM timestep, both directions (dir = blockIdx.x>>4).
// Each block owns 32 hidden units. Each of the 8 waves owns one 16-wide gate
// column tile and both 16-row batch tiles: per k-step 1 B-frag is shared by
// 2 WMMAs. Gate pre-activations from G are added, nonlinearities applied,
// c updated in place, h double-buffered by step parity.
// ---------------------------------------------------------------------------
__global__ __launch_bounds__(256) void k_lstm_step(
    const float* __restrict__ G,     // [2][S*B][G4]
    const bf16_t* __restrict__ Whh,  // [2][G4][HD]  (bf16, row-major = Bt)
    bf16_t* __restrict__ hbuf,       // [2 parity][2 dir][B][HD]
    float* __restrict__ cstate,      // [2 dir][B][HD]
    bf16_t* __restrict__ hout,       // [S][B][2*HD]
    int step, int pin)
{
    __shared__ float gsm[32][132];   // [b][gate*32 + j], padded

    const int dir   = (int)(blockIdx.x >> 4);
    const int jbase = ((int)blockIdx.x & 15) * 32;
    const int s     = dir ? (S_DIM - 1 - step) : step;
    const int wave  = threadIdx.x >> 5;      // = column tile ct (0..7)
    const int lane  = threadIdx.x & 31;
    const int l15   = lane & 15;
    const int hi    = lane >> 4;

    const bf16_t* hin = hbuf + ((size_t)pin * 2 + dir) * (B_DIM * HD);
    const bf16_t* Wd  = Whh + (size_t)dir * G4 * HD;

    const int gate = wave >> 1;
    const int cc   = (wave & 1) * 16;
    const int n    = gate * HD + jbase + cc + l15;   // Whh row (= Bt row)

    const bf16_t* Ap0 = hin + (size_t)l15 * HD + hi * 16;         // rows 0-15
    const bf16_t* Ap1 = Ap0 + (size_t)16 * HD;                   // rows 16-31
    const bf16_t* Bp  = Wd + (size_t)n * HD + hi * 16;

    v8f acc0 = {}, acc1 = {};
#pragma unroll 4
    for (int k = 0; k < HD; k += 32) {
        v16bf bv = *(const v16bf*)(Bp + k);
        v16bf a0 = ld_afrag(Ap0 + k);
        v16bf a1 = ld_afrag(Ap1 + k);
        acc0 = __builtin_amdgcn_wmma_f32_16x16x32_bf16(false, a0, false, bv, (short)0, acc0, false, false);
        acc1 = __builtin_amdgcn_wmma_f32_16x16x32_bf16(false, a1, false, bv, (short)0, acc1, false, false);
    }
    {
        const int lc = wave * 16 + l15;
#pragma unroll
        for (int r = 0; r < 8; ++r) gsm[hi * 8 + r][lc]      = acc0[r];
#pragma unroll
        for (int r = 0; r < 8; ++r) gsm[16 + hi * 8 + r][lc] = acc1[r];
    }
    __syncthreads();

    const float* Gd = G + ((size_t)dir * S_DIM * B_DIM + (size_t)s * B_DIM) * G4;
    float*  cd  = cstate + (size_t)dir * B_DIM * HD;
    bf16_t* hod = hbuf + ((size_t)(pin ^ 1) * 2 + dir) * (B_DIM * HD);

    for (int e = threadIdx.x; e < B_DIM * 32; e += 256) {
        const int b  = e >> 5;
        const int j  = e & 31;
        const int jg = jbase + j;
        const float* gb = Gd + (size_t)b * G4;
        float gi = gsm[b][j]      + gb[jg];
        float gf = gsm[b][32 + j] + gb[HD + jg];
        float gg = gsm[b][64 + j] + gb[2 * HD + jg];
        float go = gsm[b][96 + j] + gb[3 * HD + jg];
        float si = 1.0f / (1.0f + __expf(-gi));
        float sf = 1.0f / (1.0f + __expf(-gf));
        float so = 1.0f / (1.0f + __expf(-go));
        float tg = tanhf(gg);
        float c  = sf * cd[b * HD + jg] + si * tg;
        float h  = so * tanhf(c);
        cd[b * HD + jg]  = c;
        hod[b * HD + jg] = (bf16_t)h;
        hout[((size_t)s * B_DIM + b) * (2 * HD) + dir * HD + jg] = (bf16_t)h;
    }
}

// ---------------------------------------------------------------------------
extern "C" void kernel_launch(void* const* d_in, const int* in_sizes, int n_in,
                              void* d_out, int out_size, void* d_ws, size_t ws_size,
                              hipStream_t stream)
{
    const float* emb  = (const float*)d_in[0];
    const float* ctx  = (const float*)d_in[1];
    const float* pemb = (const float*)d_in[2];
    const float* Wih0 = (const float*)d_in[3];
    const float* Whh0 = (const float*)d_in[4];
    const float* bih0 = (const float*)d_in[5];
    const float* bhh0 = (const float*)d_in[6];
    const float* Wih1 = (const float*)d_in[7];
    const float* Whh1 = (const float*)d_in[8];
    const float* bih1 = (const float*)d_in[9];
    const float* bhh1 = (const float*)d_in[10];
    const float* W1   = (const float*)d_in[11];
    const float* b1   = (const float*)d_in[12];
    const float* W2   = (const float*)d_in[13];
    const float* b2   = (const float*)d_in[14];
    const int*   ids  = (const int*)d_in[15];
    const int*   pidx = (const int*)d_in[16];
    float* out = (float*)d_out;

    // ---- workspace bump allocator (d_ws is 256B aligned from hipMalloc) ----
    char* ws = (char*)d_ws;
    size_t off = 0;
    auto alloc = [&](size_t bytes) -> char* {
        char* p = ws + off;
        off += (bytes + 255) & ~(size_t)255;
        return p;
    };
    const size_t MROWS = (size_t)S_DIM * B_DIM;              // 8192
    int*    map     = (int*)    alloc((size_t)B_DIM * S_DIM * 4);
    bf16_t* allb    = (bf16_t*) alloc(MROWS * E_DIM * 2);
    bf16_t* wih0_bf = (bf16_t*) alloc((size_t)2 * G4 * E_DIM * 2);
    bf16_t* whh0_bf = (bf16_t*) alloc((size_t)2 * G4 * HD * 2);
    bf16_t* wih1_bf = (bf16_t*) alloc((size_t)2 * G4 * H_DIM * 2);
    bf16_t* whh1_bf = (bf16_t*) alloc((size_t)2 * G4 * HD * 2);
    bf16_t* w1_bf   = (bf16_t*) alloc((size_t)H_DIM * H_DIM * 2);
    bf16_t* w2_bf   = (bf16_t*) alloc((size_t)E_DIM * H_DIM * 2);
    float*  bias0   = (float*)  alloc((size_t)2 * G4 * 4);
    float*  bias1   = (float*)  alloc((size_t)2 * G4 * 4);
    float*  G       = (float*)  alloc((size_t)2 * MROWS * G4 * 4);   // reused per layer
    bf16_t* h0out   = (bf16_t*) alloc(MROWS * H_DIM * 2);
    bf16_t* h1out   = (bf16_t*) alloc(MROWS * H_DIM * 2);
    bf16_t* act     = (bf16_t*) alloc(MROWS * H_DIM * 2);
    float*  mlpout  = (float*)  alloc(MROWS * E_DIM * 4);
    bf16_t* hbuf    = (bf16_t*) alloc((size_t)2 * 2 * B_DIM * HD * 2);
    float*  cstate  = (float*)  alloc((size_t)2 * B_DIM * HD * 4);

    auto conv = [&](const float* s, bf16_t* d, size_t n) {
        k_f32_to_bf16<<<(unsigned)((n + 255) / 256), 256, 0, stream>>>(s, d, (int)n);
    };

    // ---- prompt map ----
    k_fill_i32<<<(B_DIM * S_DIM + 255) / 256, 256, 0, stream>>>(map, -1, B_DIM * S_DIM);
    k_scatter_map<<<(B_DIM * P_DIM + 255) / 256, 256, 0, stream>>>(pidx, map, B_DIM * P_DIM);

    // ---- weight conversions + combined biases ----
    conv(Wih0, wih0_bf, (size_t)2 * G4 * E_DIM);
    conv(Whh0, whh0_bf, (size_t)2 * G4 * HD);
    conv(Wih1, wih1_bf, (size_t)2 * G4 * H_DIM);
    conv(Whh1, whh1_bf, (size_t)2 * G4 * HD);
    conv(W1, w1_bf, (size_t)H_DIM * H_DIM);
    conv(W2, w2_bf, (size_t)E_DIM * H_DIM);
    k_combine_bias<<<(2 * G4 + 255) / 256, 256, 0, stream>>>(bih0, bhh0, bias0, 2 * G4);
    k_combine_bias<<<(2 * G4 + 255) / 256, 256, 0, stream>>>(bih1, bhh1, bias1, 2 * G4);

    // ---- embeddings: raw -> d_out, allb (bf16, [S,B,E]) ----
    {
        long long n = (long long)B_DIM * S_DIM * E_DIM;
        k_embed<<<(unsigned)((n + 255) / 256), 256, 0, stream>>>(
            emb, ctx, pemb, ids, map, out, allb, n);
    }

    const unsigned gridG  = (unsigned)(((MROWS >> 5) * (G4 >> 5)) / 8);     // N=2048
    const unsigned gridH  = (unsigned)(((MROWS >> 5) * (H_DIM >> 5)) / 8);  // N=1024
    const size_t   Gdir   = MROWS * (size_t)G4;
    const int      stateW = 2 * B_DIM * HD;   // u32 words: hbuf(bf16 x2 par) & cstate(f32)

    // ---- 2 biLSTM layers ----
    for (int layer = 0; layer < 2; ++layer) {
        const bf16_t* Xin  = layer ? h0out : allb;
        const bf16_t* wih  = layer ? wih1_bf : wih0_bf;
        const bf16_t* whh  = layer ? whh1_bf : whh0_bf;
        const float*  bias = layer ? bias1 : bias0;
        bf16_t*       hout = layer ? h1out : h0out;
        const int     Kin  = layer ? H_DIM : E_DIM;   // both 1024

        // input-projection GEMMs (both directions), bias folded in
        for (int dir = 0; dir < 2; ++dir) {
            k_gemm_bf16<<<gridG, 256, 0, stream>>>(
                Xin, wih + (size_t)dir * G4 * Kin, bias + (size_t)dir * G4,
                G + (size_t)dir * Gdir, nullptr,
                (int)MROWS, G4, Kin, 0);
        }
        // zero h (both parities) and c
        k_zero_u32<<<(stateW + 255) / 256, 256, 0, stream>>>((uint32_t*)hbuf, stateW);
        k_zero_u32<<<(stateW + 255) / 256, 256, 0, stream>>>((uint32_t*)cstate, stateW);
        // recurrence: 256 sequential steps, 32 blocks each (2 dirs x 16 slices)
        for (int st = 0; st < S_DIM; ++st) {
            k_lstm_step<<<32, 256, 0, stream>>>(G, whh, hbuf, cstate, hout, st, st & 1);
        }
    }

    // ---- output MLP ----
    k_gemm_bf16<<<gridH, 256, 0, stream>>>(h1out, w1_bf, b1, nullptr, act,
                                           (int)MROWS, H_DIM, H_DIM, 1);
    k_gemm_bf16<<<gridH, 256, 0, stream>>>(act, w2_bf, b2, mlpout, nullptr,
                                           (int)MROWS, E_DIM, H_DIM, 0);

    // ---- scatter prompt-position updates into d_out ----
    {
        int n = B_DIM * P_DIM * E_DIM;
        k_scatter_out<<<(n + 255) / 256, 256, 0, stream>>>(mlpout, pemb, pidx, out, n);
    }
    (void)in_sizes; (void)n_in; (void)out_size; (void)ws_size;
}
